// Model_5738076307714
// MI455X (gfx1250) — matrix operand order repaired and checked by reading, not compiled, not run on hardware
//
#include <hip/hip_runtime.h>
#include <math.h>

#define SEQL   512
#define NB     32
#define DM     512
#define NH     8
#define EH     64
#define DFFN   2048
#define NLAYER 3
#define PREDL  96
#define CIN    32
#define NM     (NB * SEQL)          // 16384 total rows

typedef __attribute__((ext_vector_type(16))) _Float16 v16h;
typedef __attribute__((ext_vector_type(4)))  _Float16 v4h;
typedef __attribute__((ext_vector_type(8)))  float    v8f;
typedef __attribute__((ext_vector_type(4)))  float    v4f;

// ---------------- wave reduction helpers (wave32) ----------------
__device__ __forceinline__ float wave_sum(float v) {
    #pragma unroll
    for (int m = 16; m > 0; m >>= 1) v += __shfl_xor(v, m, 32);
    return v;
}
__device__ __forceinline__ float wave_max(float v) {
    #pragma unroll
    for (int m = 16; m > 0; m >>= 1) v = fmaxf(v, __shfl_xor(v, m, 32));
    return v;
}

// ---------------- WMMA tiled GEMM, register-pipelined ----------------
// C[M,N] = A[M,K] (row-major, lda) x op(B) + bias[n]
//   BKN=false: B stored [N,K] row-major (ldb) -> C = A * B^T   (linear layers)
//   BKN=true : B stored [K,N] row-major (ldb) -> C = A * B     (attnP*V, time proj)
// EPI: 0 = none; 1 = bias; 2 = bias + exact GELU
// GUARD: bounds-check C stores (only needed for ragged head GEMMs)
// Batch: z = blockIdx.z -> (b = z / Hsplit, h = z % Hsplit).
// Block tile 128x64, BLOCK_K = 32; 4 waves; per wave: 32 rows x 64 cols via
// 2 A-fragments x 4 B-fragments = 8x v_wmma_f32_16x16x32_f16 per K-step
// (each B fragment reused for two WMMAs from registers).
// Every loaded tile must be full (ragged operands are zero-padded host-side).
template<bool BKN, int EPI, bool GUARD>
__global__ __launch_bounds__(128) void gemm_wmma(
    const float* __restrict__ A, const float* __restrict__ Bm,
    const float* __restrict__ bias, float* __restrict__ C,
    int M, int N, int K, int lda, int ldb, int ldc,
    int Hsplit, long long sAb, long long sAh, long long sBb, long long sBh,
    long long sCb, long long sCh)
{
    __shared__ _Float16 As[128 * 36];                      // [m][k], stride 36
    __shared__ _Float16 Bs[BKN ? 32 * 68 : 64 * 36];       // [k][n]/68 or [n][k]/36

    const int z  = blockIdx.z;
    const int bb = z / Hsplit;
    const int hh = z - bb * Hsplit;
    A  += (long long)bb * sAb + (long long)hh * sAh;
    Bm += (long long)bb * sBb + (long long)hh * sBh;
    C  += (long long)bb * sCb + (long long)hh * sCh;

    const int tid    = threadIdx.x;
    const int lane   = tid & 31;
    const int wv     = tid >> 5;
    const int blockM = blockIdx.y * 128;
    const int blockN = blockIdx.x * 64;

    v8f acc[8];                                    // [mt*4 + nt]
    #pragma unroll
    for (int t = 0; t < 8; ++t)
        acc[t] = (v8f){0.f, 0.f, 0.f, 0.f, 0.f, 0.f, 0.f, 0.f};

    const int mloc0 = wv * 32 + (lane & 15);       // first of two 16-row A frags
    const int hi    = lane >> 4;

    // per-thread staging coordinates (branch-free, vectorized float4 -> 4xf16)
    const int ar = (tid >> 3);          // 0..15   row within 16-row group
    const int ac = (tid & 7) << 2;      // 0,4,..28 col
    const int bn = (tid & 15) << 2;     // BKN: 0,4,..60 col
    const int bk = (tid >> 4);          // BKN: 0..7   row within 8-row group

    const float* pa = A + (long long)(blockM + ar) * lda + ac;
    const float* pb = BKN ? (Bm + (long long)bk * ldb + (blockN + bn))
                          : (Bm + (long long)(blockN + ar) * ldb + ac);

    v4f fa[8], fb[4];
    auto load_tiles = [&](int kofs) {
        #pragma unroll
        for (int i = 0; i < 8; ++i)
            fa[i] = *(const v4f*)(pa + (long long)i * 16 * lda + kofs);
        if constexpr (!BKN) {
            #pragma unroll
            for (int i = 0; i < 4; ++i)
                fb[i] = *(const v4f*)(pb + (long long)i * 16 * ldb + kofs);
        } else {
            #pragma unroll
            for (int i = 0; i < 4; ++i)
                fb[i] = *(const v4f*)(pb + (long long)(kofs + i * 8) * ldb);
        }
    };

    load_tiles(0);

    for (int kk = 0; kk < K; kk += 32) {
        // ---- commit staged registers to LDS (f32 -> f16) ----
        #pragma unroll
        for (int i = 0; i < 8; ++i)
            *(v4h*)&As[(i * 16 + ar) * 36 + ac] = __builtin_convertvector(fa[i], v4h);
        if constexpr (!BKN) {
            #pragma unroll
            for (int i = 0; i < 4; ++i)
                *(v4h*)&Bs[(i * 16 + ar) * 36 + ac] = __builtin_convertvector(fb[i], v4h);
        } else {
            #pragma unroll
            for (int i = 0; i < 4; ++i)
                *(v4h*)&Bs[(i * 8 + bk) * 68 + bn] = __builtin_convertvector(fb[i], v4h);
        }
        __syncthreads();

        // ---- issue next tile's global loads (overlap with WMMA below) ----
        if (kk + 32 < K) load_tiles(kk + 32);
        // pull the tile after that into GL2 (global_prefetch_b8, no counters)
        if (kk + 64 < K) {
            __builtin_prefetch(pa + kk + 64, 0, 1);
            if constexpr (!BKN) __builtin_prefetch(pb + kk + 64, 0, 1);
            else                __builtin_prefetch(pb + (long long)(kk + 64) * ldb, 0, 1);
        }

        // ---- 2 A fragments: lane m = lane&15; K halves split by lane-half ----
        union { v16h v; _Float16 h[16]; } af[2];
        #pragma unroll
        for (int mt = 0; mt < 2; ++mt) {
            int mrow = mloc0 + mt * 16;
            #pragma unroll
            for (int j = 0; j < 8; ++j) {
                int kb = ((j & 3) << 1) + (hi << 3) + ((j >> 2) << 4);
                af[mt].h[2 * j]     = As[mrow * 36 + kb];
                af[mt].h[2 * j + 1] = As[mrow * 36 + kb + 1];
            }
        }
        // ---- 4 B fragments (N sub-tiles), each reused for 2 WMMAs ----
        #pragma unroll
        for (int nt = 0; nt < 4; ++nt) {
            union { v16h v; _Float16 h[16]; } bf;
            int nloc = nt * 16 + (lane & 15);
            #pragma unroll
            for (int j = 0; j < 8; ++j) {
                int kb = ((j & 3) << 1) + (hi << 3) + ((j >> 2) << 4);
                if constexpr (!BKN) {
                    bf.h[2 * j]     = Bs[nloc * 36 + kb];
                    bf.h[2 * j + 1] = Bs[nloc * 36 + kb + 1];
                } else {
                    bf.h[2 * j]     = Bs[kb * 68 + nloc];
                    bf.h[2 * j + 1] = Bs[(kb + 1) * 68 + nloc];
                }
            }
            acc[nt] = __builtin_amdgcn_wmma_f32_16x16x32_f16(
                false, af[0].v, false, bf.v, (short)0, acc[nt], false, false);
            acc[4 + nt] = __builtin_amdgcn_wmma_f32_16x16x32_f16(
                false, af[1].v, false, bf.v, (short)0, acc[4 + nt], false, false);
        }
        __syncthreads();
    }

    // ---- epilogue + store (C/D layout: VGPR r -> M=r+8*hi, lane&15 -> N) ----
    #pragma unroll
    for (int nt = 0; nt < 4; ++nt) {
        int gn = blockN + nt * 16 + (lane & 15);
        if (GUARD && gn >= N) continue;
        float bval = (EPI >= 1) ? bias[gn] : 0.f;
        #pragma unroll
        for (int mt = 0; mt < 2; ++mt) {
            #pragma unroll
            for (int r = 0; r < 8; ++r) {
                int gm = blockM + wv * 32 + mt * 16 + r + hi * 8;
                if (GUARD && gm >= M) continue;
                float x = acc[mt * 4 + nt][r] + bval;
                if (EPI == 2) x = 0.5f * x * (1.0f + erff(x * 0.70710678118f));
                C[(long long)gm * ldc + gn] = x;
            }
        }
    }
}

// ---------------- instance norm over time (per b,c) ----------------
__global__ void instnorm_kernel(const float* __restrict__ x, float* __restrict__ xn,
                                float* __restrict__ meanb, float* __restrict__ stdb)
{
    int w = blockIdx.x * 4 + (threadIdx.x >> 5);     // 0..1023 (b*32+c)
    int lane = threadIdx.x & 31;
    if (w >= NB * CIN) return;
    int b = w >> 5, c = w & 31;
    const float* p = x + (long long)b * SEQL * CIN + c;
    float v[16], s = 0.f;
    #pragma unroll
    for (int i = 0; i < 16; ++i) { v[i] = p[(lane + i * 32) * CIN]; s += v[i]; }
    float mean = wave_sum(s) * (1.f / SEQL);
    float q = 0.f;
    #pragma unroll
    for (int i = 0; i < 16; ++i) { float d = v[i] - mean; q += d * d; }
    float stdev = sqrtf(wave_sum(q) * (1.f / SEQL) + 1e-5f);
    float inv = 1.f / stdev;
    float* po = xn + (long long)b * SEQL * CIN + c;
    #pragma unroll
    for (int i = 0; i < 16; ++i) po[(lane + i * 32) * CIN] = (v[i] - mean) * inv;
    if (lane == 0) { meanb[w] = mean; stdb[w] = stdev; }
}

// ---------------- circular conv1d(k=3) embedding + sinusoidal PE ----------------
__global__ void embed_kernel(const float* __restrict__ xn,
                             const float* __restrict__ conv_w,
                             float* __restrict__ h)
{
    __shared__ float xr[3][CIN];
    int blk = blockIdx.x;                  // b*SEQL + l
    int b = blk >> 9, l = blk & (SEQL - 1);
    int tid = threadIdx.x;
    if (tid < 96) {
        int k = tid >> 5, c = tid & 31;
        int t = (l + k - 1 + SEQL) & (SEQL - 1);
        xr[k][c] = xn[((long long)b * SEQL + t) * CIN + c];
    }
    __syncthreads();
    for (int d = tid; d < DM; d += 256) {
        const float* w = conv_w + (long long)d * CIN * 3;   // [D][C][3]
        float s = 0.f;
        #pragma unroll 8
        for (int c = 0; c < CIN; ++c)
            s += xr[0][c] * w[c * 3] + xr[1][c] * w[c * 3 + 1] + xr[2][c] * w[c * 3 + 2];
        int i2 = d & ~1;
        float div = __expf(-(float)i2 * 0.01798894658f);    // ln(10000)/512
        float ang = (float)l * div;
        s += (d & 1) ? __cosf(ang) : __sinf(ang);
        h[(long long)blk * DM + d] = s;
    }
}

// ---------------- scaled softmax over rows of length 512 ----------------
__global__ void softmax_kernel(float* __restrict__ S, int nrows)
{
    int row = blockIdx.x * 4 + (threadIdx.x >> 5);
    if (row >= nrows) return;
    int lane = threadIdx.x & 31;
    float* p = S + (long long)row * SEQL;
    union { v4f v[4]; float f[16]; } u;
    #pragma unroll
    for (int i = 0; i < 4; ++i)
        u.v[i] = *(const v4f*)(p + i * 128 + lane * 4) * 0.125f;
    float mx = -3.4e38f;
    #pragma unroll
    for (int i = 0; i < 16; ++i) mx = fmaxf(mx, u.f[i]);
    mx = wave_max(mx);
    float s = 0.f;
    #pragma unroll
    for (int i = 0; i < 16; ++i) { u.f[i] = __expf(u.f[i] - mx); s += u.f[i]; }
    float inv = 1.f / wave_sum(s);
    #pragma unroll
    for (int i = 0; i < 4; ++i)
        *(v4f*)(p + i * 128 + lane * 4) = u.v[i] * inv;
}

// ---------------- layernorm over D=512 with optional residual ----------------
__global__ void layernorm_kernel(const float* __restrict__ X, const float* __restrict__ R,
                                 const float* __restrict__ g, const float* __restrict__ bta,
                                 float* __restrict__ Y, int nrows)
{
    int row = blockIdx.x * 4 + (threadIdx.x >> 5);
    if (row >= nrows) return;
    int lane = threadIdx.x & 31;
    const float* px = X + (long long)row * DM;
    const float* pr = R ? R + (long long)row * DM : nullptr;
    union { v4f v[4]; float f[16]; } u;
    #pragma unroll
    for (int i = 0; i < 4; ++i) {
        v4f a = *(const v4f*)(px + i * 128 + lane * 4);
        if (R) a += *(const v4f*)(pr + i * 128 + lane * 4);
        u.v[i] = a;
    }
    float s = 0.f;
    #pragma unroll
    for (int i = 0; i < 16; ++i) s += u.f[i];
    float mean = wave_sum(s) * (1.f / DM);
    float q = 0.f;
    #pragma unroll
    for (int i = 0; i < 16; ++i) { float d = u.f[i] - mean; q += d * d; }
    float rstd = rsqrtf(wave_sum(q) * (1.f / DM) + 1e-5f);
    float* py = Y + (long long)row * DM;
    #pragma unroll
    for (int i = 0; i < 4; ++i) {
        int c = i * 128 + lane * 4;
        v4f gg = *(const v4f*)(g + c);
        v4f bb = *(const v4f*)(bta + c);
        *(v4f*)(py + c) = (u.v[i] - mean) * rstd * gg + bb;
    }
}

// ---------------- de-normalize head output ----------------
__global__ void head_out_kernel(const float* __restrict__ tp, const float* __restrict__ bt,
                                const float* __restrict__ meanb, const float* __restrict__ stdb,
                                float* __restrict__ out)
{
    int i = blockIdx.x * 256 + threadIdx.x;
    if (i >= NB * PREDL * CIN) return;
    int c = i & 31;
    int p = (i >> 5) % PREDL;
    int b = i / (PREDL * CIN);
    float v = tp[i] + bt[p];
    out[i] = v * stdb[b * CIN + c] + meanb[b * CIN + c];
}

extern "C" void kernel_launch(void* const* d_in, const int* in_sizes, int n_in,
                              void* d_out, int out_size, void* d_ws, size_t ws_size,
                              hipStream_t stream)
{
    (void)in_sizes; (void)n_in; (void)out_size; (void)ws_size;
    const float* x_enc  = (const float*)d_in[0];
    const float* conv_w = (const float*)d_in[1];
    const float* Wq     = (const float*)d_in[2];
    const float* Wk     = (const float*)d_in[3];
    const float* Wv     = (const float*)d_in[4];
    const float* Wo     = (const float*)d_in[5];
    const float* bq     = (const float*)d_in[6];
    const float* bk     = (const float*)d_in[7];
    const float* bv     = (const float*)d_in[8];
    const float* bo     = (const float*)d_in[9];
    const float* ff1w   = (const float*)d_in[10];
    const float* ff1b   = (const float*)d_in[11];
    const float* ff2w   = (const float*)d_in[12];
    const float* ff2b   = (const float*)d_in[13];
    const float* n1g    = (const float*)d_in[14];
    const float* n1b    = (const float*)d_in[15];
    const float* n2g    = (const float*)d_in[16];
    const float* n2b    = (const float*)d_in[17];
    const float* fng    = (const float*)d_in[18];
    const float* fnb    = (const float*)d_in[19];
    const float* Wch    = (const float*)d_in[20];
    const float* bch    = (const float*)d_in[21];
    const float* Wt     = (const float*)d_in[22];
    const float* bt     = (const float*)d_in[23];

    float* W = (float*)d_ws;
    size_t off = 0;
    auto alloc = [&](size_t n) { float* p = W + off; off += n; return p; };
    float* xn    = alloc((size_t)NB * SEQL * CIN);
    float* meanb = alloc(NB * CIN);
    float* stdb  = alloc(NB * CIN);
    float* h    = alloc((size_t)NM * DM);
    float* qb   = alloc((size_t)NM * DM);
    float* kb   = alloc((size_t)NM * DM);
    float* vb   = alloc((size_t)NM * DM);
    float* ob   = alloc((size_t)NM * DM);
    float* x1   = alloc((size_t)NM * DM);
    float* sg   = alloc((size_t)8388608);       // shared scores-chunk / FFN-chunk buffer
    float* decp = alloc((size_t)NM * 64);       // dec padded to 64 cols (zero-filled)
    float* Wchp = alloc((size_t)64 * DM);       // Wch padded 32->64 rows
    float* Wtp  = alloc((size_t)128 * SEQL);    // Wt padded 96->128 rows
    float* tp   = alloc((size_t)NB * PREDL * CIN);

    // zero-pad the ragged operands so every GEMM tile load is full
    hipMemsetAsync(decp, 0, (size_t)NM * 64 * sizeof(float), stream);
    hipMemsetAsync(Wchp, 0, (size_t)64 * DM * sizeof(float), stream);
    hipMemcpyAsync(Wchp, Wch, (size_t)CIN * DM * sizeof(float),
                   hipMemcpyDeviceToDevice, stream);
    hipMemsetAsync(Wtp, 0, (size_t)128 * SEQL * sizeof(float), stream);
    hipMemcpyAsync(Wtp, Wt, (size_t)PREDL * SEQL * sizeof(float),
                   hipMemcpyDeviceToDevice, stream);

    instnorm_kernel<<<256, 128, 0, stream>>>(x_enc, xn, meanb, stdb);
    embed_kernel<<<NB * SEQL, 256, 0, stream>>>(xn, conv_w, h);

    for (int i = 0; i < NLAYER; ++i) {
        const float* Wq_i = Wq + (size_t)i * DM * DM;
        const float* Wk_i = Wk + (size_t)i * DM * DM;
        const float* Wv_i = Wv + (size_t)i * DM * DM;
        const float* Wo_i = Wo + (size_t)i * DM * DM;
        const float* f1w  = ff1w + (size_t)i * DFFN * DM;
        const float* f2w  = ff2w + (size_t)i * DM * DFFN;

        // --- Q, K, V projections: [16384,512] x [512,512]^T ---
        gemm_wmma<false, 1, false><<<dim3(DM / 64, NM / 128, 1), 128, 0, stream>>>(
            h, Wq_i, bq + (size_t)i * DM, qb, NM, DM, DM, DM, DM, DM, 1, 0, 0, 0, 0, 0, 0);
        gemm_wmma<false, 1, false><<<dim3(DM / 64, NM / 128, 1), 128, 0, stream>>>(
            h, Wk_i, bk + (size_t)i * DM, kb, NM, DM, DM, DM, DM, DM, 1, 0, 0, 0, 0, 0, 0);
        gemm_wmma<false, 1, false><<<dim3(DM / 64, NM / 128, 1), 128, 0, stream>>>(
            h, Wv_i, bv + (size_t)i * DM, vb, NM, DM, DM, DM, DM, DM, 1, 0, 0, 0, 0, 0, 0);

        // --- attention, 8 chunks of 4 batches (scores buffer 33.5 MB) ---
        for (int cb = 0; cb < 8; ++cb) {
            const float* qc = qb + (size_t)cb * 4 * SEQL * DM;
            const float* kc = kb + (size_t)cb * 4 * SEQL * DM;
            const float* vc = vb + (size_t)cb * 4 * SEQL * DM;
            float*       oc = ob + (size_t)cb * 4 * SEQL * DM;
            // scores[b,h] = Q[b,:,h*64:] * K[b,:,h*64:]^T    (z = b_in*8 + h)
            gemm_wmma<false, 0, false><<<dim3(SEQL / 64, SEQL / 128, 32), 128, 0, stream>>>(
                qc, kc, nullptr, sg, SEQL, SEQL, EH, DM, DM, SEQL,
                NH, (long long)SEQL * DM, EH, (long long)SEQL * DM, EH,
                (long long)NH * SEQL * SEQL, (long long)SEQL * SEQL);
            softmax_kernel<<<(4 * NH * SEQL) / 4, 128, 0, stream>>>(sg, 4 * NH * SEQL);
            // O[b,:,h*64:] = P[b,h] * V[b,:,h*64:]   (B is [K,N] with ldb = D)
            gemm_wmma<true, 0, false><<<dim3(1, SEQL / 128, 32), 128, 0, stream>>>(
                sg, vc, nullptr, oc, SEQL, EH, SEQL, SEQL, DM, DM,
                NH, (long long)NH * SEQL * SEQL, (long long)SEQL * SEQL,
                (long long)SEQL * DM, EH, (long long)SEQL * DM, EH);
        }

        // --- output projection + residual LN1 ---
        gemm_wmma<false, 1, false><<<dim3(DM / 64, NM / 128, 1), 128, 0, stream>>>(
            ob, Wo_i, bo + (size_t)i * DM, x1, NM, DM, DM, DM, DM, DM, 1, 0, 0, 0, 0, 0, 0);
        layernorm_kernel<<<NM / 4, 128, 0, stream>>>(
            h, x1, n1g + (size_t)i * DM, n1b + (size_t)i * DM, x1, NM);

        // --- FFN in 4 row-chunks of 4096 (shared 33.5 MB hidden buffer) ---
        for (int mc = 0; mc < 4; ++mc) {
            const float* xc = x1 + (size_t)mc * 4096 * DM;
            float*       yc = ob + (size_t)mc * 4096 * DM;
            gemm_wmma<false, 2, false><<<dim3(DFFN / 64, 4096 / 128, 1), 128, 0, stream>>>(
                xc, f1w, ff1b + (size_t)i * DFFN, sg, 4096, DFFN, DM, DM, DM, DFFN,
                1, 0, 0, 0, 0, 0, 0);
            gemm_wmma<false, 1, false><<<dim3(DM / 64, 4096 / 128, 1), 128, 0, stream>>>(
                sg, f2w, ff2b + (size_t)i * DM, yc, 4096, DM, DFFN, DFFN, DFFN, DM,
                1, 0, 0, 0, 0, 0, 0);
        }
        layernorm_kernel<<<NM / 4, 128, 0, stream>>>(
            x1, ob, n2g + (size_t)i * DM, n2b + (size_t)i * DM, h, NM);
    }

    // --- final norm + channel head + time projection + de-normalize ---
    layernorm_kernel<<<NM / 4, 128, 0, stream>>>(h, nullptr, fng, fnb, h, NM);
    // dec (padded to 64-col pitch): [16384,512] x Wchp[64,512]^T, store cols < 32
    gemm_wmma<false, 1, true><<<dim3(1, NM / 128, 1), 128, 0, stream>>>(
        h, Wchp, bch, decp, NM, CIN, DM, DM, DM, 64, 1, 0, 0, 0, 0, 0, 0);
    // tp[b] = Wtp[128,512] * decp[b][512,64]   (A shared across batch, B is [K,N])
    gemm_wmma<true, 0, true><<<dim3(1, 1, 32), 128, 0, stream>>>(
        Wtp, decp, nullptr, tp, PREDL, CIN, SEQL, SEQL, 64, CIN,
        1, 0, 0, (long long)SEQL * 64, 0, (long long)PREDL * CIN, 0);
    head_out_kernel<<<(NB * PREDL * CIN + 255) / 256, 256, 0, stream>>>(
        tp, bt, meanb, stdb, (float*)d_out);
}
